// IWAE_41772851921256
// MI455X (gfx1250) — matrix operand order repaired and checked by reading, not compiled, not run on hardware
//
#include <hip/hip_runtime.h>
#include <hip/hip_bf16.h>

// ---------------------------------------------------------------------------
// VRNN / IWAE forward on gfx1250 (MI455X).
// All GEMMs run through v_wmma_f32_16x16x32_f16 (f16 in, f32 accumulate).
// Weights are converted + pre-transposed to f16 K-major once per launch; each lane's
// WMMA B fragment is 16 k rows of one column, read with stride ldb.
// ---------------------------------------------------------------------------

typedef __attribute__((ext_vector_type(16))) _Float16 v16h;
typedef __attribute__((ext_vector_type(8)))  _Float16 v8h;
typedef __attribute__((ext_vector_type(8)))  float    v8f;

#define TSTEPS 128
#define BATCH  64
#define PART   8
#define ROWS   512      // BATCH*PART
#define HDIM   512
#define ZDIM   128
#define XDIM   88
#define XPAD   96       // XDIM padded to mult of 32

#define CAT16(lo, hi) __builtin_shufflevector((lo), (hi), 0,1,2,3,4,5,6,7,8,9,10,11,12,13,14,15)

// ---------------------------------------------------------------------------
// Fused WMMA GEMM:  C[m,n] = act( sum_k A[m,k] * Bt[k,n] + bias[n] )
// A = up to 3 row-major f16 segments concatenated along K (K1/K2/K3, each a
// multiple of 32).  Bt = f16 K-major pre-transposed weight (ldb mult of 128).
// rowShift: A row index = (m >> rowShift)  (handles jnp.repeat(x_t, P)).
// Per wave: one 16-row M tile x four 16-col N tiles.  Block: 8 waves = 64x128.
// Inner loop: segment-unrolled, pointer-increment addressing, double-buffered
// fragments so loads never overwrite registers the in-flight WMMA reads.
// act: 0 none, 1 relu, 2 sigmoid, 3 softplus.
// ---------------------------------------------------------------------------
__launch_bounds__(256)
__global__ void vrnn_gemm_wmma(
    const _Float16* __restrict__ A1, int lda1,
    const _Float16* __restrict__ A2, int lda2,
    const _Float16* __restrict__ A3, int lda3,
    int K1, int K2, int K3, int rowShift,
    const _Float16* __restrict__ Bt, int ldb,
    const float* __restrict__ bias, int Nact,
    float* __restrict__ Cf, int ldcf,
    _Float16* __restrict__ Ch, int ldch,
    int act)
{
    const int lane = threadIdx.x & 31;
    const int wave = threadIdx.x >> 5;
    const int wm   = wave & 3;            // 4 waves along M
    const int wn   = wave >> 2;           // 2 waves along N
    const int m0   = blockIdx.y * 64 + wm * 16;
    const int n0   = blockIdx.x * 128 + wn * 64;
    const int mrow = lane & 15;
    const int hi   = lane >> 4;           // half-wave select
    const int arow = (m0 + mrow) >> rowShift;
    const size_t bstep = (size_t)32 * ldb;

    v8f acc[4] = {};

    // running B pointer: column (n0 + lane&15), k rows in table order, advanced by 32 rows per chunk
    const _Float16* bp = Bt + n0 + mrow;
    auto ldB = [&](const _Float16* p) { v16h r; for (int e = 0; e < 16; ++e) r[e] = p[(size_t)((e & 7) + (hi << 3) + ((e >> 3) << 4)) * ldb]; return r; };

    #pragma unroll
    for (int s = 0; s < 3; ++s) {
        const int segK = (s == 0) ? K1 : (s == 1) ? K2 : K3;
        if (segK <= 0) continue;
        const _Float16* __restrict__ ab  = (s == 0) ? A1 : (s == 1) ? A2 : A3;
        const int                    lda = (s == 0) ? lda1 : (s == 1) ? lda2 : lda3;

        // A fragment (ISA 16-bit 16x32 layout):
        //   lane<16 : M=lane,    halves = K{0..7, 16..23}
        //   lane>=16: M=lane-16, halves = K{8..15, 24..31}
        const _Float16* ap = ab + (size_t)arow * lda + (hi << 3);

        // ---- prologue: load first chunk's fragments
        v8h  alo = *(const v8h*)(ap);
        v8h  ahi = *(const v8h*)(ap + 16);
        v16h b0  = ldB(bp);
        v16h b1  = ldB(bp + 16);
        v16h b2  = ldB(bp + 32);
        v16h b3  = ldB(bp + 48);

        for (int k = 32; k < segK; k += 32) {
            // ---- load next chunk into the alternate register set
            const _Float16* apn = ap + k;
            const _Float16* bpn = bp + (size_t)k * ldb;
            __builtin_prefetch(bpn + bstep, 0, 3);   // WGP-scope prefetch
            v8h  alo_n = *(const v8h*)(apn);
            v8h  ahi_n = *(const v8h*)(apn + 16);
            v16h c0    = ldB(bpn);
            v16h c1    = ldB(bpn + 16);
            v16h c2    = ldB(bpn + 32);
            v16h c3    = ldB(bpn + 48);

            // ---- MAC current chunk
            v16h a = CAT16(alo, ahi);
            acc[0] = __builtin_amdgcn_wmma_f32_16x16x32_f16(false, a, false, b0, (short)0, acc[0], false, false);
            acc[1] = __builtin_amdgcn_wmma_f32_16x16x32_f16(false, a, false, b1, (short)0, acc[1], false, false);
            acc[2] = __builtin_amdgcn_wmma_f32_16x16x32_f16(false, a, false, b2, (short)0, acc[2], false, false);
            acc[3] = __builtin_amdgcn_wmma_f32_16x16x32_f16(false, a, false, b3, (short)0, acc[3], false, false);

            // ---- rotate buffers
            alo = alo_n; ahi = ahi_n;
            b0 = c0; b1 = c1; b2 = c2; b3 = c3;
        }

        // ---- epilogue chunk of this segment
        {
            v16h a = CAT16(alo, ahi);
            acc[0] = __builtin_amdgcn_wmma_f32_16x16x32_f16(false, a, false, b0, (short)0, acc[0], false, false);
            acc[1] = __builtin_amdgcn_wmma_f32_16x16x32_f16(false, a, false, b1, (short)0, acc[1], false, false);
            acc[2] = __builtin_amdgcn_wmma_f32_16x16x32_f16(false, a, false, b2, (short)0, acc[2], false, false);
            acc[3] = __builtin_amdgcn_wmma_f32_16x16x32_f16(false, a, false, b3, (short)0, acc[3], false, false);
        }
        bp += (size_t)segK * ldb;
    }

    // --- epilogue: C layout VGPR j -> row (j + hi*8), col = lane&15
    const int cmBase = m0 + hi * 8;
    #pragma unroll
    for (int t = 0; t < 4; ++t) {
        int n = n0 + t * 16 + mrow;
        if (n >= Nact) continue;
        float bv = bias ? bias[n] : 0.f;
        #pragma unroll
        for (int j = 0; j < 8; ++j) {
            float v = acc[t][j] + bv;
            if      (act == 1) v = fmaxf(v, 0.f);
            else if (act == 2) v = 1.f / (1.f + __expf(-v));
            else if (act == 3) v = (v > 20.f) ? v : log1pf(__expf(v));
            int m = cmBase + j;
            if (Cf) Cf[(size_t)m * ldcf + n] = v;
            if (Ch) Ch[(size_t)m * ldch + n] = (_Float16)v;
        }
    }
}

// ---------------------------------------------------------------------------
// Prep kernels (run once per launch; fully rewrite their outputs)
// ---------------------------------------------------------------------------
__global__ void vrnn_wtrans(const float* __restrict__ W, _Float16* __restrict__ Bt,
                            int N, int K, int Kpad, int ldb)
{
    int idx = blockIdx.x * blockDim.x + threadIdx.x;
    int total = Kpad * ldb;
    if (idx >= total) return;
    int k = idx / ldb, n = idx - k * ldb;
    float v = (k < K && n < N) ? W[(size_t)n * K + k] : 0.f;
    Bt[idx] = (_Float16)v;
}

__global__ void vrnn_lstmtrans(const float* __restrict__ wih,
                               const float* __restrict__ whh,
                               _Float16* __restrict__ Bt) // 1536 x 2048
{
    int idx = blockIdx.x * blockDim.x + threadIdx.x;
    if (idx >= 1536 * 2048) return;
    int k = idx >> 11, n = idx & 2047;
    float v = (k < 1024) ? wih[(size_t)n * 1024 + k]
                         : whh[(size_t)n * 512 + (k - 1024)];
    Bt[idx] = (_Float16)v;
}

__global__ void vrnn_xconv(const float* __restrict__ x, _Float16* __restrict__ x16)
{
    int idx = blockIdx.x * blockDim.x + threadIdx.x;
    if (idx >= TSTEPS * BATCH * XPAD) return;
    int r = idx / XPAD, d = idx - r * XPAD;
    x16[idx] = (_Float16)((d < XDIM) ? x[(size_t)r * XDIM + d] : 0.f);
}

__global__ void vrnn_init(_Float16* __restrict__ h16, float* __restrict__ c,
                          float* __restrict__ lhp, float* __restrict__ kla)
{
    int i = blockIdx.x * blockDim.x + threadIdx.x;
    if (i < ROWS * HDIM) { h16[i] = (_Float16)0.f; c[i] = 0.f; }
    if (i < BATCH) { lhp[i] = 0.f; kla[i] = 0.f; }
}

// ---------------------------------------------------------------------------
// Per-timestep elementwise / reduction kernels
// ---------------------------------------------------------------------------
__global__ void vrnn_zk(const float* __restrict__ em, const float* __restrict__ es,
                        const float* __restrict__ epst, _Float16* __restrict__ z16)
{
    int i = blockIdx.x * blockDim.x + threadIdx.x;
    if (i >= ROWS * ZDIM) return;
    z16[i] = (_Float16)(em[i] + es[i] * epst[i]);
}

__global__ void vrnn_lstm_ew(const float* __restrict__ gates,
                             float* __restrict__ c, _Float16* __restrict__ h16)
{
    int i = blockIdx.x * blockDim.x + threadIdx.x;
    if (i >= ROWS * HDIM) return;
    int r = i >> 9, cc = i & 511;                       // HDIM == 512
    const float* g = gates + ((size_t)r << 11);         // 4*HDIM == 2048
    float iv = g[cc], fv = g[cc + 512], gv = g[cc + 1024], ov = g[cc + 1536];
    float si = 1.f / (1.f + __expf(-iv));
    float sf = 1.f / (1.f + __expf(-fv));
    float so = 1.f / (1.f + __expf(-ov));
    float cn = sf * c[i] + si * tanhf(gv);
    c[i] = cn;
    h16[i] = (_Float16)(so * tanhf(cn));
}

// one block per row: KL over ZDIM, NLL over XDIM
__global__ void vrnn_rowred(const float* __restrict__ em, const float* __restrict__ es,
                            const float* __restrict__ pm, const float* __restrict__ ps,
                            const float* __restrict__ dm, const float* __restrict__ xt,
                            float* __restrict__ klrow, float* __restrict__ nllrow)
{
    __shared__ float s1[128], s2[128];
    int r = blockIdx.x, th = threadIdx.x;
    float kl = 0.f;
    for (int d = th; d < ZDIM; d += 128) {
        float e = es[(size_t)r * ZDIM + d];
        float p = ps[(size_t)r * ZDIM + d];
        float dd = em[(size_t)r * ZDIM + d] - pm[(size_t)r * ZDIM + d];
        kl += 2.f * logf(p) - 2.f * logf(e) + (e * e + dd * dd) / (p * p) - 1.f;
    }
    float nll = 0.f;
    const float* xr = xt + (size_t)(r >> 3) * XDIM;     // PART == 8
    for (int d = th; d < XDIM; d += 128) {
        float xv = xr[d], dv = dm[(size_t)r * XDIM + d];
        nll -= xv * logf(dv) + (1.f - xv) * logf(1.f - dv);
    }
    s1[th] = kl; s2[th] = nll;
    __syncthreads();
    for (int off = 64; off > 0; off >>= 1) {
        if (th < off) { s1[th] += s1[th + off]; s2[th] += s2[th + off]; }
        __syncthreads();
    }
    if (th == 0) { klrow[r] = 0.5f * s1[0]; nllrow[r] = s2[0]; }
}

// one block, 64 threads (one per batch element): logsumexp over particles
__global__ void vrnn_combine(const float* __restrict__ klrow,
                             const float* __restrict__ nllrow,
                             const float* __restrict__ mt,
                             float* __restrict__ lhp_acc, float* __restrict__ kl_acc)
{
    __shared__ float sm[64];
    int b = threadIdx.x;
    float m = mt[b];
    float kls = 0.f, mx = -1e30f;
    float la[PART];
    for (int p = 0; p < PART; ++p) {
        float kv = klrow[b * PART + p];
        kls += kv;
        float l = -(nllrow[b * PART + p] + kv) * m;
        la[p] = l;
        mx = fmaxf(mx, l);
    }
    float s = 0.f;
    for (int p = 0; p < PART; ++p) s += __expf(la[p] - mx);
    float lse = mx + logf(s);
    lhp_acc[b] += (lse - logf((float)PART)) * m;
    sm[b] = kls;
    __syncthreads();
    for (int off = 32; off > 0; off >>= 1) {
        if (b < off) sm[b] += sm[b + off];
        __syncthreads();
    }
    kl_acc[b] += (sm[0] / (float)(ROWS)) * m;   // kl.mean() over all B*P rows
}

__global__ void vrnn_final(const float* __restrict__ lhp, const float* __restrict__ kla,
                           float* __restrict__ out)
{
    __shared__ float sm[64];
    int b = threadIdx.x;
    sm[b] = lhp[b];
    __syncthreads();
    for (int off = 32; off > 0; off >>= 1) {
        if (b < off) sm[b] += sm[b + off];
        __syncthreads();
    }
    if (b == 0) out[0] = -sm[0];
    out[1 + b]  = lhp[b];
    out[65 + b] = kla[b];
}

// ---------------------------------------------------------------------------
// Host side
// ---------------------------------------------------------------------------
static inline size_t alignup(size_t x) { return (x + 255) & ~(size_t)255; }

static void launch_gemm(hipStream_t s,
    const _Float16* A1, int lda1, const _Float16* A2, int lda2,
    const _Float16* A3, int lda3, int K1, int K2, int K3, int rowShift,
    const _Float16* Bt, int ldb, const float* bias, int Nact,
    float* Cf, int ldcf, _Float16* Ch, int ldch, int act)
{
    dim3 grid(ldb / 128, ROWS / 64);
    vrnn_gemm_wmma<<<grid, 256, 0, s>>>(A1, lda1, A2, lda2, A3, lda3,
                                        K1, K2, K3, rowShift, Bt, ldb,
                                        bias, Nact, Cf, ldcf, Ch, ldch, act);
}

extern "C" void kernel_launch(void* const* d_in, const int* in_sizes, int n_in,
                              void* d_out, int out_size, void* d_ws, size_t ws_size,
                              hipStream_t stream)
{
    (void)in_sizes; (void)n_in; (void)out_size; (void)ws_size;

    const float* x    = (const float*)d_in[0];   // (T,B,88)
    const float* mask = (const float*)d_in[1];   // (T,B)
    const float* eps  = (const float*)d_in[2];   // (T,B*P,128)
    // d_in[3] = num_particles (==8, hard-coded)
    const float* W_phix1  = (const float*)d_in[4];   const float* b_phix1  = (const float*)d_in[5];
    const float* W_phix2  = (const float*)d_in[6];   const float* b_phix2  = (const float*)d_in[7];
    const float* W_phiz   = (const float*)d_in[8];   const float* b_phiz   = (const float*)d_in[9];
    const float* W_enc1   = (const float*)d_in[10];  const float* b_enc1   = (const float*)d_in[11];
    const float* W_enc2   = (const float*)d_in[12];  const float* b_enc2   = (const float*)d_in[13];
    const float* W_encm   = (const float*)d_in[14];  const float* b_encm   = (const float*)d_in[15];
    const float* W_encs   = (const float*)d_in[16];  const float* b_encs   = (const float*)d_in[17];
    const float* W_prior  = (const float*)d_in[18];  const float* b_prior  = (const float*)d_in[19];
    const float* W_prm    = (const float*)d_in[20];  const float* b_prm    = (const float*)d_in[21];
    const float* W_prs    = (const float*)d_in[22];  const float* b_prs    = (const float*)d_in[23];
    const float* W_dec1   = (const float*)d_in[24];  const float* b_dec1   = (const float*)d_in[25];
    const float* W_dec2   = (const float*)d_in[26];  const float* b_dec2   = (const float*)d_in[27];
    const float* W_decm   = (const float*)d_in[28];  const float* b_decm   = (const float*)d_in[29];
    const float* W_wih    = (const float*)d_in[30];
    const float* W_whh    = (const float*)d_in[31];

    // ---- workspace bump allocator ----
    char* base = (char*)d_ws;
    size_t off = 0;
    auto allocB = [&](size_t bytes) -> void* {
        void* p = base + off; off = alignup(off + bytes); return p;
    };
    auto allocH = [&](size_t n) { return (_Float16*)allocB(n * sizeof(_Float16)); };
    auto allocF = [&](size_t n) { return (float*)allocB(n * sizeof(float)); };

    // f16 pre-transposed weights (Kpad x ldb)
    _Float16* Bt_phix1 = allocH((size_t)XPAD * 512);
    _Float16* Bt_phix2 = allocH((size_t)512  * 512);
    _Float16* Bt_phiz  = allocH((size_t)128  * 512);
    _Float16* Bt_enc1  = allocH((size_t)1024 * 512);
    _Float16* Bt_enc2  = allocH((size_t)512  * 512);
    _Float16* Bt_encm  = allocH((size_t)512  * 128);
    _Float16* Bt_encs  = allocH((size_t)512  * 128);
    _Float16* Bt_prior = allocH((size_t)512  * 512);
    _Float16* Bt_prm   = allocH((size_t)512  * 128);
    _Float16* Bt_prs   = allocH((size_t)512  * 128);
    _Float16* Bt_dec1  = allocH((size_t)1024 * 512);
    _Float16* Bt_dec2  = allocH((size_t)512  * 512);
    _Float16* Bt_decm  = allocH((size_t)512  * 128);
    _Float16* Bt_lstm  = allocH((size_t)1536 * 2048);
    // f16 activations
    _Float16* x16   = allocH((size_t)TSTEPS * BATCH * XPAD);
    _Float16* tmpA  = allocH((size_t)ROWS * HDIM);
    _Float16* tmpB  = allocH((size_t)ROWS * HDIM);
    _Float16* phx16 = allocH((size_t)ROWS * HDIM);
    _Float16* phz16 = allocH((size_t)ROWS * HDIM);
    _Float16* h16   = allocH((size_t)ROWS * HDIM);
    _Float16* ench  = allocH((size_t)ROWS * HDIM);
    _Float16* pr16  = allocH((size_t)ROWS * HDIM);
    _Float16* z16   = allocH((size_t)ROWS * ZDIM);
    // f32 buffers
    float* em    = allocF((size_t)ROWS * ZDIM);
    float* es    = allocF((size_t)ROWS * ZDIM);
    float* pm    = allocF((size_t)ROWS * ZDIM);
    float* ps    = allocF((size_t)ROWS * ZDIM);
    float* dm    = allocF((size_t)ROWS * XDIM);
    float* gates = allocF((size_t)ROWS * 2048);
    float* cbuf  = allocF((size_t)ROWS * HDIM);
    float* klrow = allocF(ROWS);
    float* nllrow= allocF(ROWS);
    float* lhp   = allocF(BATCH);
    float* kla   = allocF(BATCH);

    // ---- prep: weight transpose+convert, x convert, state init ----
    auto wt = [&](const float* W, _Float16* Bt, int N, int K, int Kpad, int ldb) {
        int total = Kpad * ldb;
        vrnn_wtrans<<<(total + 255) / 256, 256, 0, stream>>>(W, Bt, N, K, Kpad, ldb);
    };
    wt(W_phix1, Bt_phix1, 512, XDIM, XPAD, 512);
    wt(W_phix2, Bt_phix2, 512, 512, 512, 512);
    wt(W_phiz,  Bt_phiz,  512, 128, 128, 512);
    wt(W_enc1,  Bt_enc1,  512, 1024, 1024, 512);
    wt(W_enc2,  Bt_enc2,  512, 512, 512, 512);
    wt(W_encm,  Bt_encm,  128, 512, 512, 128);
    wt(W_encs,  Bt_encs,  128, 512, 512, 128);
    wt(W_prior, Bt_prior, 512, 512, 512, 512);
    wt(W_prm,   Bt_prm,   128, 512, 512, 128);
    wt(W_prs,   Bt_prs,   128, 512, 512, 128);
    wt(W_dec1,  Bt_dec1,  512, 1024, 1024, 512);
    wt(W_dec2,  Bt_dec2,  512, 512, 512, 512);
    wt(W_decm,  Bt_decm,  XDIM, 512, 512, 128);
    vrnn_lstmtrans<<<(1536 * 2048 + 255) / 256, 256, 0, stream>>>(W_wih, W_whh, Bt_lstm);
    vrnn_xconv<<<(TSTEPS * BATCH * XPAD + 255) / 256, 256, 0, stream>>>(x, x16);
    vrnn_init<<<(ROWS * HDIM + 255) / 256, 256, 0, stream>>>(h16, cbuf, lhp, kla);

    const _Float16* NH = nullptr;
    // ---- sequential scan over timesteps ----
    for (int t = 0; t < TSTEPS; ++t) {
        const _Float16* xt16 = x16 + (size_t)t * BATCH * XPAD;
        const float*    xt   = x    + (size_t)t * BATCH * XDIM;
        const float*    mt   = mask + (size_t)t * BATCH;
        const float*    epst = eps  + (size_t)t * ROWS * ZDIM;

        // phx = relu(phi_x2 @ relu(phi_x1 @ repeat(x_t, P)))
        launch_gemm(stream, xt16, XPAD, NH, 0, NH, 0, XPAD, 0, 0, 3,
                    Bt_phix1, 512, b_phix1, 512, nullptr, 0, tmpA, 512, 1);
        launch_gemm(stream, tmpA, 512, NH, 0, NH, 0, 512, 0, 0, 0,
                    Bt_phix2, 512, b_phix2, 512, nullptr, 0, phx16, 512, 1);
        // enc_h = relu(enc2 @ relu(enc1 @ [phx|h]))
        launch_gemm(stream, phx16, 512, h16, 512, NH, 0, 512, 512, 0, 0,
                    Bt_enc1, 512, b_enc1, 512, nullptr, 0, tmpA, 512, 1);
        launch_gemm(stream, tmpA, 512, NH, 0, NH, 0, 512, 0, 0, 0,
                    Bt_enc2, 512, b_enc2, 512, nullptr, 0, ench, 512, 1);
        // em, es
        launch_gemm(stream, ench, 512, NH, 0, NH, 0, 512, 0, 0, 0,
                    Bt_encm, 128, b_encm, 128, em, 128, nullptr, 0, 0);
        launch_gemm(stream, ench, 512, NH, 0, NH, 0, 512, 0, 0, 0,
                    Bt_encs, 128, b_encs, 128, es, 128, nullptr, 0, 3);
        // prior branch
        launch_gemm(stream, h16, 512, NH, 0, NH, 0, 512, 0, 0, 0,
                    Bt_prior, 512, b_prior, 512, nullptr, 0, pr16, 512, 1);
        launch_gemm(stream, pr16, 512, NH, 0, NH, 0, 512, 0, 0, 0,
                    Bt_prm, 128, b_prm, 128, pm, 128, nullptr, 0, 0);
        launch_gemm(stream, pr16, 512, NH, 0, NH, 0, 512, 0, 0, 0,
                    Bt_prs, 128, b_prs, 128, ps, 128, nullptr, 0, 3);
        // z = em + es * eps  (rsample)
        vrnn_zk<<<(ROWS * ZDIM + 255) / 256, 256, 0, stream>>>(em, es, epst, z16);
        // phz = relu(phi_z @ z)
        launch_gemm(stream, z16, ZDIM, NH, 0, NH, 0, ZDIM, 0, 0, 0,
                    Bt_phiz, 512, b_phiz, 512, nullptr, 0, phz16, 512, 1);
        // dec_h = relu(dec2 @ relu(dec1 @ [phz|h]))
        launch_gemm(stream, phz16, 512, h16, 512, NH, 0, 512, 512, 0, 0,
                    Bt_dec1, 512, b_dec1, 512, nullptr, 0, tmpA, 512, 1);
        launch_gemm(stream, tmpA, 512, NH, 0, NH, 0, 512, 0, 0, 0,
                    Bt_dec2, 512, b_dec2, 512, nullptr, 0, tmpB, 512, 1);
        // dm = sigmoid(dec_mean @ dec_h)
        launch_gemm(stream, tmpB, 512, NH, 0, NH, 0, 512, 0, 0, 0,
                    Bt_decm, 128, b_decm, XDIM, dm, XDIM, nullptr, 0, 2);
        // gates = [phx|phz|h] @ [Wih|Whh]^T  (one fused K=1536, N=2048 GEMM)
        launch_gemm(stream, phx16, 512, phz16, 512, h16, 512, 512, 512, 512, 0,
                    Bt_lstm, 2048, nullptr, 2048, gates, 2048, nullptr, 0, 0);
        // LSTM cell update (c f32, h f16)
        vrnn_lstm_ew<<<(ROWS * HDIM + 255) / 256, 256, 0, stream>>>(gates, cbuf, h16);
        // per-row KL / NLL, then per-batch logsumexp + accumulators
        vrnn_rowred<<<ROWS, 128, 0, stream>>>(em, es, pm, ps, dm, xt, klrow, nllrow);
        vrnn_combine<<<1, 64, 0, stream>>>(klrow, nllrow, mt, lhp, kla);
    }

    // out = [-sum(lhp), lhp(64), kla(64)]  -> 129 floats
    vrnn_final<<<1, 64, 0, stream>>>(lhp, kla, (float*)d_out);
}
